// Decoder_51986284150954
// MI455X (gfx1250) — matrix operand order repaired and checked by reading, not compiled, not run on hardware
//
#include <hip/hip_runtime.h>

// ---------------------------------------------------------------------------
// CDNA5 (gfx1250) decoder: autoregressive GRU -> LSTM -> FC.
// bf16 WMMA (V_WMMA_F32_16X16X32_BF16) with fp32 accumulation.
// Weights repacked into *step-major* fragment blocks so each k-step's B tiles
// are one contiguous 4-6KB chunk, staged per block into an LDS double buffer
// via GLOBAL_LOAD_ASYNC_TO_LDS_B128 (ASYNCcnt) overlapping the matrix pipe.
// ---------------------------------------------------------------------------

typedef __attribute__((ext_vector_type(16))) __bf16 v16bf;
typedef __attribute__((ext_vector_type(8)))  __bf16 v8bf;
typedef __attribute__((ext_vector_type(8)))  float  v8f;
typedef int v4i __attribute__((vector_size(16)));   // matches async builtin param

#define BATCH 128
#define HDIM  512
#define LDIM  1024
#define VDIM  1024
#define TSEQ  256

// fp32 -> bf16, round to nearest even
__device__ __forceinline__ __bf16 f2bf(float f) {
    union { float f; unsigned u; } in; in.f = f;
    unsigned u = in.u;
    u += 0x7FFFu + ((u >> 16) & 1u);
    union { unsigned short s; __bf16 b; } out; out.s = (unsigned short)(u >> 16);
    return out.b;
}

__device__ __forceinline__ float sigm(float x)  { return 1.0f / (1.0f + __expf(-x)); }
__device__ __forceinline__ float tanh_(float x) { return 2.0f * sigm(2.0f * x) - 1.0f; }

// --- async global->LDS 16B copy (GLOBAL_LOAD_ASYNC_TO_LDS_B128, ASYNCcnt) ---
__device__ __forceinline__ void async_copy16(const void* g, void* l) {
#if __has_builtin(__builtin_amdgcn_global_load_async_to_lds_b128)
    void* gnc = const_cast<void*>(g);
    __builtin_amdgcn_global_load_async_to_lds_b128(
        (__attribute__((address_space(1))) v4i*)gnc,
        (__attribute__((address_space(3))) v4i*)l, 0, 0);
#else
    *(v4i*)l = *(const v4i*)g;           // fallback: through VGPRs
#endif
}

__device__ __forceinline__ void wait_async() {
#if __has_builtin(__builtin_amdgcn_s_wait_asynccnt)
    __builtin_amdgcn_s_wait_asynccnt(0);
#else
    asm volatile("s_wait_asynccnt 0" ::: "memory");
#endif
}

// A-fragment (16x32 bf16, MxK) in this file's contiguous k order (B packed to match):
//   lanes 0-15 : row = lane,    K = {k0+0..15}
//   lanes 16-31: row = lane-16, K = {k0+16..31}
__device__ __forceinline__ v16bf load_a(const __bf16* __restrict__ rowbase, int k0, int lane) {
    const __bf16* p = rowbase + k0 + ((lane & 16) ? 16 : 0);
    v8bf lo = *(const v8bf*)(p);
    v8bf hi = *(const v8bf*)(p + 8);
    return __builtin_shufflevector(lo, hi, 0,1,2,3,4,5,6,7,8,9,10,11,12,13,14,15);
}

// B-fragment read from an LDS-staged tile (32 lanes x 16 bf16 = 1KB/tile)
__device__ __forceinline__ v16bf load_b_lds(const char* sh, int frag, int lane) {
    return *(const v16bf*)(sh + frag * 1024 + lane * 32);
}

__device__ __forceinline__ v8f wmma_bf16(v16bf a, v16bf b, v8f c) {
    return __builtin_amdgcn_wmma_f32_16x16x32_bf16(false, a, false, b, (short)0, c, false, false);
}

// ---------------------------------------------------------------------------
// Weight packing, step-major: the fragments a block consumes in one k-step are
// contiguous.  Fragment = 32 lanes x 16 bf16 (1KB); B[k,n] = W[n,k].
//   lane l<16 : n = ntile*16 + l, K = kt*32 + j        (j = 0..15)
//   lane l>=16: n = ntile*16 + (l-16), K = kt*32 + 16 + j
// ---------------------------------------------------------------------------

// GRU: P[((nt*16 + kt)*6 + f)*512 + lane*16 + j];  f<3: gate f of Wih, else Whh
__global__ void pack_gru_kernel(const float* __restrict__ Wih,
                                const float* __restrict__ Whh,
                                __bf16* __restrict__ P) {
    long idx  = (long)blockIdx.x * 256 + threadIdx.x;   // < 32*16*6*512
    int  j    = (int)(idx & 15);
    int  lane = (int)((idx >> 4) & 31);
    int  f    = (int)((idx >> 9) % 6);
    long tk   = (idx >> 9) / 6;          // nt*16 + kt
    int  kt   = (int)(tk & 15);
    int  nt   = (int)(tk >> 4);
    int  g    = (f < 3) ? f : f - 3;
    const float* W = (f < 3) ? Wih : Whh;
    int n = g * HDIM + nt * 16 + (lane & 15);
    int k = kt * 32 + ((lane & 16) ? 16 : 0) + j;
    P[idx] = f2bf(W[(long)n * HDIM + k]);
}

// LSTM: P[((nt*48 + kk)*4 + f)*512 + lane*16 + j];  kk<16 -> Wih(K=512), else Whh(K=1024)
__global__ void pack_lstm_kernel(const float* __restrict__ Wih,
                                 const float* __restrict__ Whh,
                                 __bf16* __restrict__ P) {
    long idx  = (long)blockIdx.x * 256 + threadIdx.x;   // < 64*48*4*512
    int  j    = (int)(idx & 15);
    int  lane = (int)((idx >> 4) & 31);
    int  f    = (int)((idx >> 9) & 3);
    long tk   = idx >> 11;               // nt*48 + kk
    int  kk   = (int)(tk % 48);
    int  nt   = (int)(tk / 48);
    int  n    = f * LDIM + nt * 16 + (lane & 15);
    int  sel  = (lane & 16) ? 16 : 0;
    float v;
    if (kk < 16) v = Wih[(long)n * HDIM + kk * 32 + sel + j];
    else         v = Whh[(long)n * LDIM + (kk - 16) * 32 + sel + j];
    P[idx] = f2bf(v);
}

// FC: P[((ng4*32 + kt)*4 + f)*512 + lane*16 + j];  n = ng4*64 + f*16 + lane%16
__global__ void pack_fc_kernel(const float* __restrict__ W,
                               __bf16* __restrict__ P) {
    long idx  = (long)blockIdx.x * 256 + threadIdx.x;   // < 16*32*4*512
    int  j    = (int)(idx & 15);
    int  lane = (int)((idx >> 4) & 31);
    int  f    = (int)((idx >> 9) & 3);
    long tk   = idx >> 11;               // ng4*32 + kt
    int  kt   = (int)(tk & 31);
    int  ng4  = (int)(tk >> 5);
    int  n    = ng4 * 64 + f * 16 + (lane & 15);
    int  k    = kt * 32 + ((lane & 16) ? 16 : 0) + j;
    P[idx] = f2bf(W[(long)n * LDIM + k]);
}

__global__ void f32_to_bf16_kernel(const float* __restrict__ src, __bf16* __restrict__ dst, int n) {
    int i = blockIdx.x * blockDim.x + threadIdx.x;
    if (i < n) dst[i] = f2bf(src[i]);
}

// ---------------------------------------------------------------------------
// GRU step. Block = 8 waves = 8 M-tiles; blockIdx = one 16-col slice of H.
// 6 shared B tiles/k-step (r,z,n x {ih,hh}) = 6KB contiguous -> LDS dbl buf.
// Grid: <<<32, 256>>>.
// ---------------------------------------------------------------------------
__global__ void gru_step_kernel(const __bf16* __restrict__ a_bf,   // [128,512]
                                const float*  __restrict__ h_src,  // [128,512]
                                const __bf16* __restrict__ Pg,     // step-major packed
                                const float* __restrict__ bih, const float* __restrict__ bhh,
                                float*  __restrict__ x_out,
                                __bf16* __restrict__ xbf_out,
                                __bf16* __restrict__ seq_t,
                                int first) {
    __shared__ __align__(32) char shB[2][6 * 1024];
    int tid  = threadIdx.x;
    int lane = tid & 31, wid = tid >> 5;
    int mt = wid, nt = blockIdx.x;         // mt 0..7, nt 0..31
    int m0 = mt * 16, n0 = nt * 16;

    auto stage = [&](int kt, int buf) {
        const char* base = (const char*)(Pg + ((long)(nt * 16 + kt) * 6) * 512);
        char* dst = shB[buf];
        async_copy16(base + tid * 16, dst + tid * 16);
        if (tid < 128)
            async_copy16(base + (tid + 256) * 16, dst + (tid + 256) * 16);
    };

    const __bf16* arow = a_bf + (long)(m0 + (lane & 15)) * HDIM;
    v8f ri = {}, zi = {}, ni = {}, rh = {}, zh = {}, nh = {};

    stage(0, 0);
    for (int kt = 0; kt < HDIM / 32; ++kt) {
        int buf = kt & 1;
        wait_async();
        __syncthreads();
        if (kt + 1 < HDIM / 32) stage(kt + 1, buf ^ 1);
        v16bf a  = load_a(arow, kt * 32, lane);
        v16bf b0 = load_b_lds(shB[buf], 0, lane);
        v16bf b1 = load_b_lds(shB[buf], 1, lane);
        v16bf b2 = load_b_lds(shB[buf], 2, lane);
        v16bf b3 = load_b_lds(shB[buf], 3, lane);
        v16bf b4 = load_b_lds(shB[buf], 4, lane);
        v16bf b5 = load_b_lds(shB[buf], 5, lane);
        ri = wmma_bf16(a, b0, ri);
        zi = wmma_bf16(a, b1, zi);
        ni = wmma_bf16(a, b2, ni);
        rh = wmma_bf16(a, b3, rh);
        zh = wmma_bf16(a, b4, zh);
        nh = wmma_bf16(a, b5, nh);
    }

    int col = n0 + (lane & 15);
    float b_ir = bih[col], b_iz = bih[HDIM + col], b_in = bih[2 * HDIM + col];
    float b_hr = bhh[col], b_hz = bhh[HDIM + col], b_hn = bhh[2 * HDIM + col];
    int mrow0 = m0 + ((lane & 16) ? 8 : 0);

#pragma unroll
    for (int e = 0; e < 8; ++e) {
        int m = mrow0 + e;
        float gir = (first ? 0.f : ri[e]) + b_ir;   // first step: x0 = 0 -> gi = bih
        float giz = (first ? 0.f : zi[e]) + b_iz;
        float gin = (first ? 0.f : ni[e]) + b_in;
        float r  = sigm(gir + rh[e] + b_hr);
        float zg = sigm(giz + zh[e] + b_hz);
        float ng = tanh_(gin + r * (nh[e] + b_hn));
        float hv = h_src[(long)m * HDIM + col];
        float xn = (1.0f - zg) * ng + zg * hv;
        long  o  = (long)m * HDIM + col;
        x_out[o]   = xn;
        xbf_out[o] = f2bf(xn);
        seq_t[o]   = f2bf(xn > 0.f ? xn : 0.f);     // relu for LSTM input
    }
}

// ---------------------------------------------------------------------------
// LSTM step. Block = 8 M-tiles; blockIdx = one 16-col slice of L (64 blocks).
// 4 shared B tiles/k-step (i,f,g,o) = 4KB contiguous; K = 48 steps.
// ---------------------------------------------------------------------------
__global__ void lstm_step_kernel(const __bf16* __restrict__ seq_t,  // [128,512]
                                 const __bf16* __restrict__ h_in,   // [128,1024]
                                 const __bf16* __restrict__ Pl,     // step-major packed
                                 const float* __restrict__ bih, const float* __restrict__ bhh,
                                 float*  __restrict__ c,            // [128,1024] in-place
                                 __bf16* __restrict__ h_out,
                                 __bf16* __restrict__ hs_t) {
    __shared__ __align__(32) char shB[2][4 * 1024];
    int tid  = threadIdx.x;
    int lane = tid & 31, wid = tid >> 5;
    int mt = wid, nt = blockIdx.x;         // mt 0..7, nt 0..63
    int m0 = mt * 16, n0 = nt * 16;

    auto stage = [&](int kk, int buf) {
        const char* base = (const char*)(Pl + ((long)(nt * 48 + kk) * 4) * 512);
        async_copy16(base + tid * 16, shB[buf] + tid * 16);
    };

    const __bf16* arow_x = seq_t + (long)(m0 + (lane & 15)) * HDIM;
    const __bf16* arow_h = h_in  + (long)(m0 + (lane & 15)) * LDIM;
    v8f ai = {}, af = {}, ag = {}, ao = {};

    stage(0, 0);
    for (int kk = 0; kk < 48; ++kk) {
        int buf = kk & 1;
        wait_async();
        __syncthreads();
        if (kk + 1 < 48) stage(kk + 1, buf ^ 1);
        v16bf a = (kk < 16) ? load_a(arow_x, kk * 32, lane)
                            : load_a(arow_h, (kk - 16) * 32, lane);
        v16bf b0 = load_b_lds(shB[buf], 0, lane);
        v16bf b1 = load_b_lds(shB[buf], 1, lane);
        v16bf b2 = load_b_lds(shB[buf], 2, lane);
        v16bf b3 = load_b_lds(shB[buf], 3, lane);
        ai = wmma_bf16(a, b0, ai);
        af = wmma_bf16(a, b1, af);
        ag = wmma_bf16(a, b2, ag);
        ao = wmma_bf16(a, b3, ao);
    }

    int col = n0 + (lane & 15);
    float bi  = bih[col]            + bhh[col];
    float bf_ = bih[LDIM + col]     + bhh[LDIM + col];
    float bg  = bih[2 * LDIM + col] + bhh[2 * LDIM + col];
    float bo  = bih[3 * LDIM + col] + bhh[3 * LDIM + col];
    int mrow0 = m0 + ((lane & 16) ? 8 : 0);

#pragma unroll
    for (int e = 0; e < 8; ++e) {
        int m = mrow0 + e;
        long o = (long)m * LDIM + col;
        float c_old = c[o];
        float cn = sigm(af[e] + bf_) * c_old + sigm(ai[e] + bi) * tanh_(ag[e] + bg);
        float hn = sigm(ao[e] + bo) * tanh_(cn);
        c[o]     = cn;
        h_out[o] = f2bf(hn);
        hs_t[o]  = f2bf(hn);
    }
}

// ---------------------------------------------------------------------------
// FC: logits[b,t,v] = hs[t,b,:] . fc_W[v,:] + fc_b[v]
// M = 32768 (rm = b*T + t), K = 1024, N = 1024. Block = 8 M-tiles, one 64-col
// group; 4 shared B tiles/k-step = 4KB contiguous. Grid: <<<4096, 256>>>.
// ---------------------------------------------------------------------------
__global__ void fc_kernel(const __bf16* __restrict__ hs,  // [T,B,L] bf16
                          const __bf16* __restrict__ Pf,  // step-major packed
                          const float* __restrict__ bias,
                          float* __restrict__ out) {      // [B,T,V]
    __shared__ __align__(32) char shB[2][4 * 1024];
    int tid  = threadIdx.x;
    int lane = tid & 31, wid = tid >> 5;
    int ng4 = blockIdx.x >> 8;                       // 0..15 (col group of 64)
    int mt  = ((blockIdx.x & 255) << 3) + wid;       // 0..2047 row-tile
    int m0  = mt * 16, n0 = ng4 * 64;

    auto stage = [&](int kt, int buf) {
        const char* base = (const char*)(Pf + ((long)(ng4 * 32 + kt) * 4) * 512);
        async_copy16(base + tid * 16, shB[buf] + tid * 16);
    };

    int rm = m0 + (lane & 15);                       // row = b*T + t
    int b  = rm >> 8;
    int t  = rm & 255;
    const __bf16* arow = hs + ((long)t * BATCH + b) * LDIM;

    v8f acc[4] = {{}, {}, {}, {}};
    stage(0, 0);
    for (int kt = 0; kt < LDIM / 32; ++kt) {
        int buf = kt & 1;
        wait_async();
        __syncthreads();
        if (kt + 1 < LDIM / 32) stage(kt + 1, buf ^ 1);
        v16bf a = load_a(arow, kt * 32, lane);
#pragma unroll
        for (int j = 0; j < 4; ++j)
            acc[j] = wmma_bf16(a, load_b_lds(shB[buf], j, lane), acc[j]);
    }

    int mrow0 = m0 + ((lane & 16) ? 8 : 0);
#pragma unroll
    for (int j = 0; j < 4; ++j) {
        int col = n0 + j * 16 + (lane & 15);
        float bv = bias[col];
#pragma unroll
        for (int e = 0; e < 8; ++e) {
            int rme = mrow0 + e;
            out[(long)rme * VDIM + col] = acc[j][e] + bv;
        }
    }
}

// ---------------------------------------------------------------------------
// Host side
// ---------------------------------------------------------------------------
extern "C" void kernel_launch(void* const* d_in, const int* in_sizes, int n_in,
                              void* d_out, int out_size, void* d_ws, size_t ws_size,
                              hipStream_t stream) {
    const float* z        = (const float*)d_in[0];
    const float* gru_Wih  = (const float*)d_in[2];
    const float* gru_Whh  = (const float*)d_in[3];
    const float* gru_bih  = (const float*)d_in[4];
    const float* gru_bhh  = (const float*)d_in[5];
    const float* lstm_Wih = (const float*)d_in[6];
    const float* lstm_Whh = (const float*)d_in[7];
    const float* lstm_bih = (const float*)d_in[8];
    const float* lstm_bhh = (const float*)d_in[9];
    const float* fc_W     = (const float*)d_in[10];
    const float* fc_b     = (const float*)d_in[11];
    float* logits = (float*)d_out;

    char* ws = (char*)d_ws;
    size_t off = 0;
    auto alloc = [&](size_t bytes) { char* p = ws + off; off += (bytes + 255) & ~(size_t)255; return p; };

    const long NG = (long)32 * 16 * 6 * 512;   // GRU packed elements
    const long NL = (long)64 * 48 * 4 * 512;   // LSTM packed elements
    const long NF = (long)16 * 32 * 4 * 512;   // FC packed elements

    __bf16* P_gru  = (__bf16*)alloc((size_t)NG * 2);
    __bf16* P_lstm = (__bf16*)alloc((size_t)NL * 2);
    __bf16* P_fc   = (__bf16*)alloc((size_t)NF * 2);
    __bf16* z_bf   = (__bf16*)alloc((size_t)BATCH * HDIM * 2);
    float*  x_f32  = (float*) alloc((size_t)2 * BATCH * HDIM * 4);
    __bf16* x_bf   = (__bf16*)alloc((size_t)2 * BATCH * HDIM * 2);
    __bf16* h_bf   = (__bf16*)alloc((size_t)2 * BATCH * LDIM * 2);
    float*  c_st   = (float*) alloc((size_t)BATCH * LDIM * 4);
    __bf16* seq_bf = (__bf16*)alloc((size_t)TSEQ * BATCH * HDIM * 2);
    __bf16* hs_bf  = (__bf16*)alloc((size_t)TSEQ * BATCH * LDIM * 2);
    (void)ws_size; (void)n_in; (void)in_sizes; (void)out_size;

    // zero-init initial states (seq[0] = relu(x0) = 0, h0 = 0, c0 = 0)
    (void)hipMemsetAsync(seq_bf, 0, (size_t)BATCH * HDIM * 2, stream);
    (void)hipMemsetAsync(h_bf,   0, (size_t)BATCH * LDIM * 2, stream);
    (void)hipMemsetAsync(c_st,   0, (size_t)BATCH * LDIM * 4, stream);

    pack_gru_kernel <<<(unsigned)(NG / 256), 256, 0, stream>>>(gru_Wih,  gru_Whh,  P_gru);
    pack_lstm_kernel<<<(unsigned)(NL / 256), 256, 0, stream>>>(lstm_Wih, lstm_Whh, P_lstm);
    pack_fc_kernel  <<<(unsigned)(NF / 256), 256, 0, stream>>>(fc_W, P_fc);

    f32_to_bf16_kernel<<<(BATCH * HDIM + 255) / 256, 256, 0, stream>>>(z, z_bf, BATCH * HDIM);

    // --- autoregressive GRU: t = 1..T-1 (seq[0] = 0 already) ---
    for (int t = 1; t < TSEQ; ++t) {
        int first = (t == 1);
        const __bf16* a    = first ? z_bf : x_bf + ((size_t)((t - 1) & 1)) * BATCH * HDIM;
        const float*  hsrc = first ? z    : x_f32 + ((size_t)((t - 1) & 1)) * BATCH * HDIM;
        gru_step_kernel<<<32, 256, 0, stream>>>(
            a, hsrc, P_gru, gru_bih, gru_bhh,
            x_f32 + ((size_t)(t & 1)) * BATCH * HDIM,
            x_bf  + ((size_t)(t & 1)) * BATCH * HDIM,
            seq_bf + (size_t)t * BATCH * HDIM,
            first);
    }

    // --- LSTM over the full sequence ---
    for (int t = 0; t < TSEQ; ++t) {
        lstm_step_kernel<<<64, 256, 0, stream>>>(
            seq_bf + (size_t)t * BATCH * HDIM,
            h_bf + ((size_t)(t & 1)) * BATCH * LDIM,
            P_lstm, lstm_bih, lstm_bhh,
            c_st,
            h_bf + ((size_t)((t + 1) & 1)) * BATCH * LDIM,
            hs_bf + (size_t)t * BATCH * LDIM);
    }

    // --- FC projection to logits [B, T, V] ---
    fc_kernel<<<4096, 256, 0, stream>>>(hs_bf, P_fc, fc_b, logits);
}
